// APNRRU_72284299591836
// MI455X (gfx1250) — matrix operand order repaired and checked by reading, not compiled, not run on hardware
//
#include <hip/hip_runtime.h>
#include <math.h>

// Problem constants (from reference)
#define BB 4096
#define SS 1024
#define HH 16
#define HA 3
#define WIN 16
#define NF 3
#define HN 16
#define KU 43      // 2H + HA + 2NF + 2
#define NO 35      // 2H + HA
#define TCH 8      // output time-chunk buffered in LDS

typedef _Float16 half_t;
typedef __attribute__((ext_vector_type(16))) _Float16 v16h;
typedef __attribute__((ext_vector_type(8)))  float    v8f;
typedef __attribute__((ext_vector_type(4)))  float    f4raw;

union AFrag { v16h v; f4raw f4[2]; _Float16 e[16]; };
union CFrag { v8f  v; float e[8]; };

// Branchless transcendentals for the serial scan:
//  - CDNA5 native V_TANH_F32 via builtin.
//  - sigmoid via v_exp + v_rcp (no IEEE-div fixup sequence, no EXEC branches).
__device__ __forceinline__ float fast_tanh(float x) {
#if __has_builtin(__builtin_amdgcn_tanhf)
    return __builtin_amdgcn_tanhf(x);
#else
    float e = __expf(-2.0f * __builtin_fabsf(x));
    float t = 1.0f - 2.0f * e * __builtin_amdgcn_rcpf(1.0f + e);
    return (x < 0.0f) ? -t : t;
#endif
}

__device__ __forceinline__ float sigmoid_f(float x) {
    return __builtin_amdgcn_rcpf(1.0f + __expf(-x));
}

// ---------------------------------------------------------------------------
// Phase 1: time-parallel features. For every (b, t):
//   rotation rr, ri; 4x FIR combos; rotated IQ(8) -> f16 workspace [b][t][8]
// ---------------------------------------------------------------------------
__global__ __launch_bounds__(256) void apn_phase1(
    const float* __restrict__ x,
    const float* __restrict__ firI, const float* __restrict__ firQ,
    half_t* __restrict__ wIQ, float* __restrict__ wrr, float* __restrict__ wri)
{
    int id = blockIdx.x * blockDim.x + threadIdx.x;
    int b = id / SS;
    int t = id % SS;
    const float* xb = x + (size_t)b * SS * 2;

    float wi[WIN], wq[WIN];
#pragma unroll
    for (int k = 0; k < WIN; ++k) {
        int idx = t + k - (WIN - 1);
        bool ok = (idx >= 0);
        wi[k] = ok ? xb[idx * 2 + 0] : 0.0f;
        wq[k] = ok ? xb[idx * 2 + 1] : 0.0f;
    }
    float xi = wi[WIN - 1], xq = wq[WIN - 1];
    float mag = sqrtf(xi * xi + xq * xq);
    float rr = xi / mag;
    float ri = -xq / mag;

    float I4[4], Q4[4];
#pragma unroll
    for (int f = 0; f < NF; ++f) {
        float fII = 0.f, fQQ = 0.f, fQI = 0.f, fIQ = 0.f;
#pragma unroll
        for (int k = 0; k < WIN; ++k) {
            float wIk = firI[f * WIN + k];
            float wQk = firQ[f * WIN + k];
            fII = fmaf(wi[k], wIk, fII);
            fQQ = fmaf(wq[k], wQk, fQQ);
            fQI = fmaf(wi[k], wQk, fQI);
            fIQ = fmaf(wq[k], wIk, fIQ);
        }
        I4[f] = fII - fQQ;
        Q4[f] = fQI + fIQ;
    }
    I4[3] = xi; Q4[3] = xq;

    union { half_t h[8]; f4raw f4; } iq;
#pragma unroll
    for (int i = 0; i < 4; ++i) {
        iq.h[2 * i + 0] = (half_t)(rr * I4[i] - ri * Q4[i]);
        iq.h[2 * i + 1] = (half_t)(ri * I4[i] + rr * Q4[i]);
    }
    size_t off = (size_t)b * SS + t;
    *(f4raw*)(wIQ + off * 8) = iq.f4;
    wrr[off] = rr;
    wri[off] = ri;
}

// ---------------------------------------------------------------------------
// Phase 2: sequential scan. One wave (32 lanes) owns a 16-batch tile and
// iterates t = 0..S-1. Per step:
//   A(16x64,f16) = [xt | a | bb | hA | 0] ; 2x v_wmma_f32_16x16x32_f16 vs W_u^T
//   v = tanh(.)  -> A'(16x32,f16, K 16..31 = 0) ; 3x wmma vs W_h^T tiles
//   hn = sigmoid(C*h_new) + Z*tanh(.) ; rotate back into state; rank-1 output.
// Global per-step inputs are software-pipelined one step ahead so their
// latency is covered by the full WMMA/VALU body of the previous step.
// ---------------------------------------------------------------------------
__global__ __launch_bounds__(32) void apn_scan(
    const half_t* __restrict__ wIQ,
    const float* __restrict__ wrr, const float* __restrict__ wri,
    const float* __restrict__ Wu, const float* __restrict__ bu,
    const float* __restrict__ Wh, const float* __restrict__ bh,
    const float* __restrict__ Cp, const float* __restrict__ Zp,
    const float* __restrict__ oIw, const float* __restrict__ oQw,
    float* __restrict__ outp)
{
    const int lane = threadIdx.x;      // 0..31
    const int n_l  = lane & 15;        // N / M-in-tile index
    const int grp  = lane >> 4;        // lane group (0: lanes 0-15, 1: 16-31)
    const int base = blockIdx.x * 16;  // batch tile base

    __shared__ float  sh_hI[16][16];
    __shared__ float  sh_hQ[16][16];
    __shared__ float  sh_hA[16][4];
    __shared__ float  sh_ab[16][36];                      // h_new in f32
    __shared__ __align__(16) half_t sh_A[16][64];         // step-input A matrix
    __shared__ __align__(16) half_t sh_v[16][16];         // v after first tanh
    __shared__ float  sh_rr[16], sh_ri[16];
    __shared__ float  sh_wI[16], sh_wQ[16];
    __shared__ __align__(16) float sh_out[16][TCH * 2];   // buffered outputs

    // ---- init state + constants --------------------------------------------
    for (int i = lane; i < 16 * 16; i += 32) {
        int m = i >> 4, j = i & 15;
        sh_hI[m][j] = 0.0f;
        sh_hQ[m][j] = 0.0f;
    }
    for (int i = lane; i < 16 * 4; i += 32) sh_hA[i >> 2][i & 3] = 0.0f;
    for (int i = lane; i < 16 * 24; i += 32) {            // zero A cols 40..63
        int m = i / 24, c = 40 + (i % 24);
        sh_A[m][c] = (half_t)0.0f;
    }
    if (lane < 16) { sh_wI[lane] = oIw[lane]; sh_wQ[lane] = oQw[lane]; }

    // B fragments: B[k][n] = W[n][k]. Layout: e[h] -> K = chunk*32 + (h<8 ? grp*8 + h : 16 + grp*8 + h-8)
    AFrag Bu0, Bu1, Bh0, Bh1, Bh2;
#pragma unroll
    for (int h = 0; h < 16; ++h) {
        int k0 = 0 * 32 + ((h < 8) ? (grp * 8 + h) : (16 + grp * 8 + (h - 8)));
        int k1 = 1 * 32 + ((h < 8) ? (grp * 8 + h) : (16 + grp * 8 + (h - 8)));
        Bu0.e[h] = (k0 < KU) ? (half_t)Wu[n_l * KU + k0] : (half_t)0.0f;
        Bu1.e[h] = (k1 < KU) ? (half_t)Wu[n_l * KU + k1] : (half_t)0.0f;
        int kh = (h < 8) ? (grp * 8 + h) : (16 + grp * 8 + (h - 8)); // valid K for 2nd GEMM: < 16
        Bh0.e[h] = (kh < 16) ? (half_t)Wh[(0 * 16 + n_l) * 16 + kh] : (half_t)0.0f;
        Bh1.e[h] = (kh < 16) ? (half_t)Wh[(1 * 16 + n_l) * 16 + kh] : (half_t)0.0f;
        Bh2.e[h] = (kh < 16 && (32 + n_l) < NO) ? (half_t)Wh[(32 + n_l) * 16 + kh]
                                                : (half_t)0.0f;
    }
    const float bu_l  = bu[n_l];
    const float bh0_l = bh[n_l];
    const float bh1_l = bh[16 + n_l];
    const float bh2_l = ((32 + n_l) < NO) ? bh[32 + n_l] : 0.0f;
    const float z0_l  = Zp[n_l];
    const float z1_l  = Zp[16 + n_l];
    const float z2_l  = ((32 + n_l) < NO) ? Zp[32 + n_l] : 0.0f;
    const float Cc    = Cp[0];

    // ---- software-pipelined per-step global inputs (lanes 0..15) -----------
    float rr_nx = 0.0f, ri_nx = 0.0f;
    f4raw iq_nx = (f4raw)(0.0f);
    if (lane < 16) {
        size_t off0 = (size_t)(base + lane) * SS;          // t = 0
        rr_nx = wrr[off0];
        ri_nx = wri[off0];
        iq_nx = *(const f4raw*)(wIQ + off0 * 8);
    }

    __syncthreads();

    // ---- sequential scan over time -----------------------------------------
    for (int t = 0; t < SS; ++t) {
        // commit this step's inputs to LDS, then immediately issue next step's
        // global loads so their latency is hidden by the step body below.
        if (lane < 16) {
            int m = lane;
            sh_rr[m] = rr_nx;
            sh_ri[m] = ri_nx;
            *(f4raw*)&sh_A[m][0] = iq_nx;
#pragma unroll
            for (int j = 0; j < HA; ++j) {
                float ha = sh_hA[m][j];
                sh_ab[m][32 + j] = ha;
                sh_A[m][40 + j] = (half_t)ha;
            }
        }
        if (t + 1 < SS) {
            if (lane < 16) {
                size_t off = (size_t)(base + lane) * SS + (t + 1);
                rr_nx = wrr[off];
                ri_nx = wri[off];
                iq_nx = *(const f4raw*)(wIQ + off * 8);
            }
        }
        __syncthreads();

        // rotate previous state: a = hI*rr - hQ*ri ; bb = hI*ri + hQ*rr
#pragma unroll
        for (int i = 0; i < 8; ++i) {
            int idx = lane + i * 32;
            int m = idx >> 4, j = idx & 15;
            float hi = sh_hI[m][j], hq = sh_hQ[m][j];
            float rrm = sh_rr[m], rim = sh_ri[m];
            float a  = hi * rrm - hq * rim;
            float bb = hi * rim + hq * rrm;
            sh_ab[m][j]      = a;
            sh_ab[m][16 + j] = bb;
            sh_A[m][8 + j]   = (half_t)a;
            sh_A[m][24 + j]  = (half_t)bb;
        }
        __syncthreads();

        // A fragments (16-bit A layout: M=n_l, K swizzled by group-of-8)
        AFrag a0, a1;
        a0.f4[0] = *(const f4raw*)&sh_A[n_l][grp * 8];
        a0.f4[1] = *(const f4raw*)&sh_A[n_l][16 + grp * 8];
        a1.f4[0] = *(const f4raw*)&sh_A[n_l][32 + grp * 8];
        a1.f4[1] = *(const f4raw*)&sh_A[n_l][48 + grp * 8];

        // GEMM 1: u @ W_u^T + b_u, K = 43 padded to 64
        CFrag accu;
#pragma unroll
        for (int r = 0; r < 8; ++r) accu.e[r] = bu_l;
        accu.v = __builtin_amdgcn_wmma_f32_16x16x32_f16(
            false, a0.v, false, Bu0.v, (short)0, accu.v, false, false);
        accu.v = __builtin_amdgcn_wmma_f32_16x16x32_f16(
            false, a1.v, false, Bu1.v, (short)0, accu.v, false, false);

        // v = tanh(.) -> LDS (C/D layout: M = r + 8*grp, N = n_l)
#pragma unroll
        for (int r = 0; r < 8; ++r)
            sh_v[r + 8 * grp][n_l] = (half_t)fast_tanh(accu.e[r]);
        __syncthreads();

        // A' fragment for GEMM 2 (K = 16 real, 16..31 zero)
        AFrag av;
        av.f4[0] = *(const f4raw*)&sh_v[n_l][grp * 8];
        av.f4[1] = (f4raw)(0.0f);

        CFrag c0, c1, c2;
#pragma unroll
        for (int r = 0; r < 8; ++r) { c0.e[r] = bh0_l; c1.e[r] = bh1_l; c2.e[r] = bh2_l; }
        c0.v = __builtin_amdgcn_wmma_f32_16x16x32_f16(
            false, av.v, false, Bh0.v, (short)0, c0.v, false, false);
        c1.v = __builtin_amdgcn_wmma_f32_16x16x32_f16(
            false, av.v, false, Bh1.v, (short)0, c1.v, false, false);
        c2.v = __builtin_amdgcn_wmma_f32_16x16x32_f16(
            false, av.v, false, Bh2.v, (short)0, c2.v, false, false);

        // hn = sigmoid(C*h_new) + Z * tanh(v2); rotate back into state
        float a2r[8], b2r[8];
#pragma unroll
        for (int r = 0; r < 8; ++r) {
            int m2 = r + 8 * grp;
            a2r[r] = sigmoid_f(Cc * sh_ab[m2][n_l])      + z0_l * fast_tanh(c0.e[r]);
            b2r[r] = sigmoid_f(Cc * sh_ab[m2][16 + n_l]) + z1_l * fast_tanh(c1.e[r]);
        }
#pragma unroll
        for (int r = 0; r < 8; ++r) {
            int m2 = r + 8 * grp;
            float rrm = sh_rr[m2], rim = sh_ri[m2];
            sh_hI[m2][n_l] = rrm * a2r[r] + rim * b2r[r];
            sh_hQ[m2][n_l] = rrm * b2r[r] - rim * a2r[r];
            if (n_l < HA)
                sh_hA[m2][n_l] =
                    sigmoid_f(Cc * sh_ab[m2][32 + n_l]) + z2_l * fast_tanh(c2.e[r]);
        }
        __syncthreads();

        // rank-1 output projection: oI = dI - dQ, oQ = dI + dQ
        {
            float dI = 0.0f, dQ = 0.0f;
#pragma unroll
            for (int j = 0; j < 16; ++j) {
                dI = fmaf(sh_hI[n_l][j], sh_wI[j], dI);
                dQ = fmaf(sh_hQ[n_l][j], sh_wQ[j], dQ);
            }
            sh_out[n_l][(t & (TCH - 1)) * 2 + grp] = grp ? (dI + dQ) : (dI - dQ);
        }

        if ((t & (TCH - 1)) == (TCH - 1)) {
            __syncthreads();
            // lane (m = n_l, half = grp) flushes 8 consecutive floats (2x b128)
            f4raw* dst = (f4raw*)(outp + (size_t)(base + n_l) * SS * 2
                                       + (size_t)(t - (TCH - 1)) * 2);
            const f4raw* src = (const f4raw*)&sh_out[n_l][0];
            dst[grp * 2 + 0] = src[grp * 2 + 0];
            dst[grp * 2 + 1] = src[grp * 2 + 1];
        }
        __syncthreads();
    }
}

// ---------------------------------------------------------------------------
// Launch. Workspace layout (needs >= 96 MB):
//   [0, 64MB)  : f16 IQ features  [B][S][8]
//   [64, 80MB) : f32 rr           [B][S]
//   [80, 96MB) : f32 ri           [B][S]
// ---------------------------------------------------------------------------
extern "C" void kernel_launch(void* const* d_in, const int* in_sizes, int n_in,
                              void* d_out, int out_size, void* d_ws, size_t ws_size,
                              hipStream_t stream) {
    const float* x    = (const float*)d_in[0];
    // d_in[1] = h_0 (zeros; reference re-initializes state to zero anyway)
    const float* firI = (const float*)d_in[2];
    const float* firQ = (const float*)d_in[3];
    const float* Wu   = (const float*)d_in[4];
    const float* bu   = (const float*)d_in[5];
    const float* Wh   = (const float*)d_in[6];
    const float* bh   = (const float*)d_in[7];
    const float* Cp   = (const float*)d_in[8];
    const float* Zp   = (const float*)d_in[9];
    const float* oIw  = (const float*)d_in[10];
    const float* oQw  = (const float*)d_in[11];
    float* outp = (float*)d_out;

    half_t* wIQ = (half_t*)d_ws;
    float*  wrr = (float*)((char*)d_ws + (size_t)BB * SS * 8 * sizeof(half_t));
    float*  wri = wrr + (size_t)BB * SS;

    apn_phase1<<<(BB * SS) / 256, 256, 0, stream>>>(x, firI, firQ, wIQ, wrr, wri);
    apn_scan<<<BB / 16, 32, 0, stream>>>(wIQ, wrr, wri, Wu, bu, Wh, bh,
                                         Cp, Zp, oIw, oQw, outp);
}
